// MultiHeadedAttention_103079215844
// MI455X (gfx1250) — compile-verified, matrix operand order checked by reading, not run on hardware
//
#include <hip/hip_runtime.h>

// ---------------------------------------------------------------------------
// Types matching CDNA5 WMMA builtin signatures
// ---------------------------------------------------------------------------
typedef __attribute__((ext_vector_type(16))) __bf16 bf16x16;
typedef __attribute__((ext_vector_type(8)))  __bf16 bf16x8;
typedef __attribute__((ext_vector_type(8)))  float  v8f;

#define WMMA_BF16(a, b, c) \
  __builtin_amdgcn_wmma_f32_16x16x32_bf16(false, (a), false, (b), (short)0, (c), false, false)

// ---------------------------------------------------------------------------
// CDNA5 async global->LDS copy (16B per lane), tracked by ASYNCcnt.
// LDS destination address = low 32 bits of the generic (aperture) pointer.
// ---------------------------------------------------------------------------
__device__ __forceinline__ void async_copy_b128(__bf16* lds, const __bf16* g) {
  unsigned loff = (unsigned)(unsigned long long)lds;
  asm volatile("global_load_async_to_lds_b128 %0, %1, off"
               :: "v"(loff), "v"(g) : "memory");
}
__device__ __forceinline__ void wait_async() {
  asm volatile("s_wait_asynccnt 0x0" ::: "memory");
}

// ---------------------------------------------------------------------------
// VALU-only 16-lane row reductions: single fused DPP op per step.
// v_max_num_f32_dpp d,d,d row_ror:n computes d = max(ror16(d,n), d)
// (all sources read before write). Rotate steps 1/2/4/8 => allreduce
// over each 16-lane DPP row; co-executes with XDL WMMA, zero LDS traffic.
// ---------------------------------------------------------------------------
__device__ __forceinline__ float rowmax16(float x) {
  asm("v_max_num_f32_dpp %0, %0, %0 row_ror:1 row_mask:0xf bank_mask:0xf bound_ctrl:1" : "+v"(x));
  asm("v_max_num_f32_dpp %0, %0, %0 row_ror:2 row_mask:0xf bank_mask:0xf bound_ctrl:1" : "+v"(x));
  asm("v_max_num_f32_dpp %0, %0, %0 row_ror:4 row_mask:0xf bank_mask:0xf bound_ctrl:1" : "+v"(x));
  asm("v_max_num_f32_dpp %0, %0, %0 row_ror:8 row_mask:0xf bank_mask:0xf bound_ctrl:1" : "+v"(x));
  return x;
}
__device__ __forceinline__ float rowsum16(float x) {
  asm("v_add_f32_dpp %0, %0, %0 row_ror:1 row_mask:0xf bank_mask:0xf bound_ctrl:1" : "+v"(x));
  asm("v_add_f32_dpp %0, %0, %0 row_ror:2 row_mask:0xf bank_mask:0xf bound_ctrl:1" : "+v"(x));
  asm("v_add_f32_dpp %0, %0, %0 row_ror:4 row_mask:0xf bank_mask:0xf bound_ctrl:1" : "+v"(x));
  asm("v_add_f32_dpp %0, %0, %0 row_ror:8 row_mask:0xf bank_mask:0xf bound_ctrl:1" : "+v"(x));
  return x;
}

// ---------------------------------------------------------------------------
// Fragment loader: 16x32 bf16 tile, row-major with given element stride.
// ISA 16-bit A-matrix layout (B via B^T-row-major == same layout):
//   lanes 0-15  : row = lane,    K = {0..7, 16..23}
//   lanes 16-31 : row = lane-16, K = {8..15, 24..31}
// Two 16-byte loads per lane (ds_load_b128 from staged LDS tiles).
// ---------------------------------------------------------------------------
__device__ __forceinline__ bf16x16 load_frag(const __bf16* src, int stride, int lane) {
  const int row  = lane & 15;
  const int koff = (lane & 16) ? 8 : 0;
  const __bf16* p = src + row * stride + koff;
  bf16x8 lo = *(const bf16x8*)(p);
  bf16x8 hi = *(const bf16x8*)(p + 16);
  bf16x16 f;
#pragma unroll
  for (int i = 0; i < 8; i++) { f[i] = lo[i]; f[i + 8] = hi[i]; }
  return f;
}

// ---------------------------------------------------------------------------
// fp32 -> bf16 conversion (8 elements / thread, 16B stores)
// ---------------------------------------------------------------------------
__global__ __launch_bounds__(256) void cvt_f32_bf16(const float* __restrict__ in,
                                                    __bf16* __restrict__ out, int n) {
  int i = (blockIdx.x * 256 + threadIdx.x) * 8;
  if (i >= n) return;
  const float4 a = *(const float4*)(in + i);
  const float4 c = *(const float4*)(in + i + 4);
  bf16x8 o;
  o[0] = (__bf16)a.x; o[1] = (__bf16)a.y; o[2] = (__bf16)a.z; o[3] = (__bf16)a.w;
  o[4] = (__bf16)c.x; o[5] = (__bf16)c.y; o[6] = (__bf16)c.z; o[7] = (__bf16)c.w;
  *(bf16x8*)(out + i) = o;
}

// ---------------------------------------------------------------------------
// GEMM: C[4096 x 1024] = scale * X[4096 x 1024] * W^T   (W row-major == B^T)
// Block: 256 threads / 8 waves; block tile 128(M) x 64(N); staged K-step 64.
// Double-buffered async global->LDS staging: issue tile k+1, compute tile k,
// s_wait_asynccnt + ONE barrier per iteration. LDS = 2 x (16K A + 8K B) = 48KB.
// MODE 0: bf16 out, head-split [B,H,S,dk]     (Q [pre-scaled 1/sqrt(dk)], K)
// MODE 1: bf16 out, transposed [B,H,dk,S]     (V -> PV B-operand)
// MODE 2: fp32 out, row-major [4096 x 1024]   (final Wo projection -> d_out)
// ---------------------------------------------------------------------------
template <int MODE>
__global__ __launch_bounds__(256) void gemm128x64(const __bf16* __restrict__ X,
                                                  const __bf16* __restrict__ W,
                                                  void* __restrict__ OutP,
                                                  float scale) {
  __shared__ __bf16 As[2][128 * 64];  // 2 x 16 KB
  __shared__ __bf16 Bs[2][64 * 64];   // 2 x  8 KB

  const int t    = threadIdx.x;
  const int lane = t & 31;
  const int wv   = t >> 5;
  const int wm   = wv >> 1;  // 0..3
  const int wn   = wv & 1;   // 0..1
  const int m0   = blockIdx.y * 128;
  const int n0   = blockIdx.x * 64;

  const int sr = t >> 3;         // staging row base (0..31)
  const int sc = (t & 7) * 8;    // staging col (16B chunk)

  auto stage = [&](int buf, int k0) {
#pragma unroll
    for (int i = 0; i < 4; i++)  // A: 128 x 64 = 1024 chunks
      async_copy_b128(&As[buf][(sr + 32 * i) * 64 + sc],
                      X + (size_t)(m0 + sr + 32 * i) * 1024 + k0 + sc);
#pragma unroll
    for (int i = 0; i < 2; i++)  // B: 64 x 64 = 512 chunks
      async_copy_b128(&Bs[buf][(sr + 32 * i) * 64 + sc],
                      W + (size_t)(n0 + sr + 32 * i) * 1024 + k0 + sc);
  };

  v8f acc[2][2] = {};

  stage(0, 0);
  wait_async();
  __syncthreads();

  int cur = 0;
  for (int k0 = 0; k0 < 1024; k0 += 64) {
    if (k0 + 64 < 1024) stage(cur ^ 1, k0 + 64);  // prefetch next tile (async)

#pragma unroll
    for (int kk = 0; kk < 2; kk++) {
      bf16x16 a0 = load_frag(&As[cur][(wm * 32 + 0)  * 64 + kk * 32], 64, lane);
      bf16x16 a1 = load_frag(&As[cur][(wm * 32 + 16) * 64 + kk * 32], 64, lane);
      bf16x16 b0 = load_frag(&Bs[cur][(wn * 32 + 0)  * 64 + kk * 32], 64, lane);
      bf16x16 b1 = load_frag(&Bs[cur][(wn * 32 + 16) * 64 + kk * 32], 64, lane);
      acc[0][0] = WMMA_BF16(a0, b0, acc[0][0]);
      acc[0][1] = WMMA_BF16(a0, b1, acc[0][1]);
      acc[1][0] = WMMA_BF16(a1, b0, acc[1][0]);
      acc[1][1] = WMMA_BF16(a1, b1, acc[1][1]);
    }

    wait_async();     // next tile's LDS writes complete (this wave)
    __syncthreads();  // all waves done reading cur + done staging cur^1
    cur ^= 1;
  }

  // Epilogue. C/D layout: lanes 0-15 VGPR r -> M=r; lanes 16-31 -> M=r+8; N=lane&15.
  const int col   = lane & 15;
  const int rbase = (lane & 16) ? 8 : 0;
#pragma unroll
  for (int i = 0; i < 2; i++)
#pragma unroll
    for (int j = 0; j < 2; j++)
#pragma unroll
      for (int r = 0; r < 8; r++) {
        const int gm = m0 + wm * 32 + i * 16 + rbase + r;
        const int gn = n0 + wn * 32 + j * 16 + col;
        const float v = acc[i][j][r] * scale;
        if (MODE == 2) {
          ((float*)OutP)[(size_t)gm * 1024 + gn] = v;
        } else {
          const int b = gm >> 11, s = gm & 2047;
          const int h = gn >> 6,  d = gn & 63;
          __bf16* O = (__bf16*)OutP;
          if (MODE == 0)
            O[(((size_t)(b << 4) + h) * 2048 + s) * 64 + d] = (__bf16)v;
          else
            O[(((size_t)(b << 4) + h) * 64 + d) * 2048 + s] = (__bf16)v;
        }
      }
}

// ---------------------------------------------------------------------------
// Flash attention per (b,h): block = 8 waves x 16 query rows = 128 rows.
// K (32x64) and V^T (64x32) tiles double-buffered via async global->LDS,
// shared by all 8 waves; one barrier per 32-key step. Online softmax in
// registers with fused-DPP row reductions. P re-staged through a private
// LDS slice as the A-fragment of the PV WMMA. Q pre-scaled by 1/sqrt(dk).
// grid = (S/128, B*H); block = 256 threads. LDS = 2*(4K+4K) + 8K = 24 KB.
// ---------------------------------------------------------------------------
__global__ __launch_bounds__(256) void attn_flash(const __bf16* __restrict__ Qp,
                                                  const __bf16* __restrict__ Kp,
                                                  const __bf16* __restrict__ Vt,
                                                  __bf16* __restrict__ AO) {
  __shared__ __bf16 Ks[2][32 * 64];  // K rows  [32 keys][64 dk]
  __shared__ __bf16 Vs[2][64 * 32];  // V^T     [64 dk ][32 keys]
  __shared__ __bf16 Ps[8][16 * 32];  // per-wave P tile

  const int t    = threadIdx.x;
  const int lane = t & 31;
  const int wv   = t >> 5;
  const int bh   = blockIdx.y;   // b*16 + h
  const int b    = bh >> 4, h = bh & 15;
  const int s0   = blockIdx.x * 128 + wv * 16;

  const __bf16* qb = Qp + ((size_t)bh * 2048 + s0) * 64;
  const __bf16* kb = Kp + (size_t)bh * 2048 * 64;
  const __bf16* vb = Vt + (size_t)bh * 64 * 2048;

  // staging coordinates (one 16B chunk per thread per tile)
  const int krow = t >> 3, kcol = (t & 7) * 8;   // K tile: 32 x 64
  const int vrow = t >> 2, vcol = (t & 3) * 8;   // V tile: 64 x 32
  auto stage = [&](int buf, int j0) {
    async_copy_b128(&Ks[buf][krow * 64 + kcol], kb + (size_t)(j0 + krow) * 64 + kcol);
    async_copy_b128(&Vs[buf][vrow * 32 + vcol], vb + (size_t)vrow * 2048 + j0 + vcol);
  };

  // Q fragments held for the whole sweep (16 rows x 64 dk, pre-scaled)
  bf16x16 aq0 = load_frag(qb,      64, lane);
  bf16x16 aq1 = load_frag(qb + 32, 64, lane);

  float mrow[8], lrow[8];
  v8f o[4] = {};
#pragma unroll
  for (int r = 0; r < 8; r++) { mrow[r] = -3.0e38f; lrow[r] = 0.f; }

  const int col   = lane & 15;
  const int rbase = (lane & 16) ? 8 : 0;
  __bf16* pw = &Ps[wv][0];

  stage(0, 0);
  wait_async();
  __syncthreads();

  int cur = 0;
  for (int j0 = 0; j0 < 2048; j0 += 32) {
    if (j0 + 32 < 2048) stage(cur ^ 1, j0 + 32);  // async prefetch of next tile

    // scores = Q . K^T : B^T row-major == K row-major [keys][dk]
    v8f sc0 = {}, sc1 = {};
    {
      bf16x16 k00 = load_frag(&Ks[cur][0],            64, lane);
      bf16x16 k01 = load_frag(&Ks[cur][32],           64, lane);
      bf16x16 k10 = load_frag(&Ks[cur][16 * 64],      64, lane);
      bf16x16 k11 = load_frag(&Ks[cur][16 * 64 + 32], 64, lane);
      sc0 = WMMA_BF16(aq0, k00, sc0);
      sc0 = WMMA_BF16(aq1, k01, sc0);
      sc1 = WMMA_BF16(aq0, k10, sc1);
      sc1 = WMMA_BF16(aq1, k11, sc1);
    }

    // online softmax; reductions are single fused DPP ops per step
#pragma unroll
    for (int r = 0; r < 8; r++) {
      const float x0 = sc0[r];
      const float x1 = sc1[r];
      const float mx = rowmax16(fmaxf(x0, x1));
      const float mn = fmaxf(mrow[r], mx);
      const float al = __expf(mrow[r] - mn);
      const float e0 = __expf(x0 - mn);
      const float e1 = __expf(x1 - mn);
      const float rs = rowsum16(e0 + e1);
      lrow[r] = lrow[r] * al + rs;
      mrow[r] = mn;
#pragma unroll
      for (int f = 0; f < 4; f++) o[f][r] *= al;
      // stage P tile (row-major 16x32 bf16) in this wave's LDS slice
      pw[(rbase + r) * 32 + col]      = (__bf16)e0;
      pw[(rbase + r) * 32 + 16 + col] = (__bf16)e1;
    }

    // O += P . V : A = P (LDS), B^T row-major == V^T tile in LDS
    bf16x16 ap = load_frag(pw, 32, lane);
#pragma unroll
    for (int f = 0; f < 4; f++) {
      bf16x16 bvf = load_frag(&Vs[cur][(f * 16) * 32], 32, lane);
      o[f] = WMMA_BF16(ap, bvf, o[f]);
    }

    wait_async();     // next K/V tile landed in LDS (this wave's ops)
    __syncthreads();  // all waves done reading cur + staging cur^1 complete
    cur ^= 1;
  }

  // normalize and write merged-head bf16 [B, S, E]
  __bf16* ob = AO + ((size_t)(b * 2048 + s0)) * 1024 + h * 64;
#pragma unroll
  for (int r = 0; r < 8; r++) {
    const float inv = 1.0f / lrow[r];
#pragma unroll
    for (int f = 0; f < 4; f++)
      ob[(size_t)(rbase + r) * 1024 + f * 16 + col] = (__bf16)(o[f][r] * inv);
  }
}

// ---------------------------------------------------------------------------
// Host-side launch
// ---------------------------------------------------------------------------
extern "C" void kernel_launch(void* const* d_in, const int* in_sizes, int n_in,
                              void* d_out, int out_size, void* d_ws, size_t ws_size,
                              hipStream_t stream) {
  (void)in_sizes; (void)n_in; (void)out_size; (void)ws_size;

  const float* q  = (const float*)d_in[0];
  const float* k  = (const float*)d_in[1];
  const float* v  = (const float*)d_in[2];
  const float* Wq = (const float*)d_in[3];
  const float* Wk = (const float*)d_in[4];
  const float* Wv = (const float*)d_in[5];
  const float* Wo = (const float*)d_in[6];

  const size_t ACT = (size_t)4096 * 1024;  // B*S x E
  const size_t WGT = (size_t)1024 * 1024;  // E x E

  char* w = (char*)d_ws;
  __bf16* qb  = (__bf16*)w; w += ACT * 2;
  __bf16* kbm = (__bf16*)w; w += ACT * 2;
  __bf16* vbm = (__bf16*)w; w += ACT * 2;
  __bf16* wqb = (__bf16*)w; w += WGT * 2;
  __bf16* wkb = (__bf16*)w; w += WGT * 2;
  __bf16* wvb = (__bf16*)w; w += WGT * 2;
  __bf16* wob = (__bf16*)w; w += WGT * 2;
  __bf16* Qp  = (__bf16*)w; w += ACT * 2;  // [B,H,S,dk] (pre-scaled)
  __bf16* Kp  = (__bf16*)w; w += ACT * 2;  // [B,H,S,dk]
  __bf16* Vt  = (__bf16*)w; w += ACT * 2;  // [B,H,dk,S]
  __bf16* AO  = (__bf16*)w; w += ACT * 2;  // [B,S,E] bf16

  // 1) convert everything to bf16
  dim3 cgA((unsigned)(ACT / (256 * 8)));
  dim3 cgW((unsigned)(WGT / (256 * 8)));
  cvt_f32_bf16<<<cgA, 256, 0, stream>>>(q,  qb,  (int)ACT);
  cvt_f32_bf16<<<cgA, 256, 0, stream>>>(k,  kbm, (int)ACT);
  cvt_f32_bf16<<<cgA, 256, 0, stream>>>(v,  vbm, (int)ACT);
  cvt_f32_bf16<<<cgW, 256, 0, stream>>>(Wq, wqb, (int)WGT);
  cvt_f32_bf16<<<cgW, 256, 0, stream>>>(Wk, wkb, (int)WGT);
  cvt_f32_bf16<<<cgW, 256, 0, stream>>>(Wv, wvb, (int)WGT);
  cvt_f32_bf16<<<cgW, 256, 0, stream>>>(Wo, wob, (int)WGT);

  // 2) Q/K/V projections (WMMA bf16, fp32 accumulate).
  //    Q is pre-scaled by 1/sqrt(dk) = 0.125 so the softmax loop skips it.
  dim3 gg(16, 32);  // N/64, M/128
  gemm128x64<0><<<gg, 256, 0, stream>>>(qb,  wqb, (void*)Qp, 0.125f);
  gemm128x64<0><<<gg, 256, 0, stream>>>(kbm, wkb, (void*)Kp, 1.0f);
  gemm128x64<1><<<gg, 256, 0, stream>>>(vbm, wvb, (void*)Vt, 1.0f);

  // 3) flash attention
  attn_flash<<<dim3(16, 32), 256, 0, stream>>>(Qp, Kp, Vt, AO);

  // 4) output projection -> fp32 d_out
  gemm128x64<2><<<gg, 256, 0, stream>>>(AO, wob, d_out, 1.0f);
}
